// CTMDNXCell_76338748719516
// MI455X (gfx1250) — matrix operand order repaired and checked by reading, not compiled, not run on hardware
//
#include <hip/hip_runtime.h>
#include <hip/hip_bf16.h>
#include <math.h>

typedef __attribute__((ext_vector_type(16))) __bf16 v16bf;
typedef __attribute__((ext_vector_type(8)))  __bf16 bf16x8;
typedef __attribute__((ext_vector_type(8)))  float  v8f;
typedef __attribute__((ext_vector_type(4)))  unsigned int v4u;
typedef __attribute__((ext_vector_type(8)))  int  v8i;
typedef __attribute__((ext_vector_type(4)))  int  v4i;

#define BATCH  32768
#define INPUT  128
#define HIDDEN 256
#define NSYNCN 256
#define BH     (BATCH * HIDDEN)          // 8388608
#define INV_BH (1.0f / 8388608.0f)
#define UNFOLDS 6

// Padded LDS row strides (elements): +8 bf16 (= 4 DWORDs) per row so that the
// 16-lane A-fragment loads stride 4 banks apart instead of 16-way conflicting.
#define LDSX 136   // 128 + 8
#define LDSH 264   // 256 + 8

// ---------------------------------------------------------------------------
// TDM: stage a 64-row x K-col bf16 tile (row-major, stride K) into LDS with
// 4-DWORD padding after every row. D# per CDNA5 ISA.
//   pad_interval code: 5 -> 64 DWORDs (K=128), 6 -> 128 DWORDs (K=256)
//   pad_amount  code: 3 -> 4 DWORDs
// This toolchain's builtin takes 6 args (g0, g1, g2, g3, g4, cpol).
// ---------------------------------------------------------------------------
__device__ __forceinline__ void tdm_load_tile(const __bf16* lds_dst,
                                              const __bf16* gsrc,
                                              int K, unsigned pad_interval_code)
{
    unsigned lds_addr = (unsigned)(size_t)lds_dst;          // low 32b = LDS byte addr
    unsigned long long ga = (unsigned long long)(size_t)gsrc;

    v4u g0;
    g0[0] = 1u;                                             // count=1, no gather
    g0[1] = lds_addr;                                       // lds_addr[31:0]
    g0[2] = (unsigned)(ga & 0xFFFFFFFFu);                   // global_addr[31:0]
    g0[3] = (unsigned)((ga >> 32) & 0x01FFFFFFu) | (2u << 30); // addr[56:32] | type=2

    v8i g1;
    unsigned w0 = (1u << 16)                                // data_size = 1 -> 2 bytes
                | (1u << 20)                                // pad_enable
                | (pad_interval_code << 22)                 // pad_interval
                | (3u << 25);                               // pad_amount = 4 DWORDs
    g1[0] = (int)w0;
    g1[1] = (int)((unsigned)K << 16);                       // tensor_dim0[15:0] in [31:16]
    g1[2] = (int)((unsigned)(BATCH & 0xFFFF) << 16);        // tensor_dim1 lo16 in [31:16]
    g1[3] = (int)((unsigned)K << 16);                       // tile_dim0 in [31:16]
    g1[4] = 64;                                             // tile_dim1 = 64 rows
    g1[5] = K;                                              // tensor_dim0_stride lo32
    g1[6] = 0;
    g1[7] = 0;

    v4i z4 = {0, 0, 0, 0};                                  // 2D tile: groups 2/3 unused
    v8i z8 = {0, 0, 0, 0, 0, 0, 0, 0};
    __builtin_amdgcn_tensor_load_to_lds(g0, g1, z4, z4, z8, 0);
}

// ---------------------------------------------------------------------------
// Fused GEMM segment: acc += A_lds[rows, K] x W(n0.., k)^T
// A in LDS (row stride LDA, padded), W row-major bf16 in global (= B^T).
// Wave computes 32(M) x 64(N); acc[ms][j] is the 16x16 subtile.
// ---------------------------------------------------------------------------
template<int K, int LDA>
__device__ __forceinline__ void gemm_seg(const __bf16* As,
                                         const __bf16* __restrict__ Wbase, int ldw,
                                         v8f acc[2][4], int lane)
{
    const int l15   = lane & 15;
    const int koffA = (lane < 16) ? 0 : 16;   // A: lower lanes K 0-15, upper K 16-31
    const int khB   = (lane < 16) ? 0 : 16;   // B: lower lanes K 0-15, upper K 16-31
    const __bf16* a0p = As + (size_t)l15 * LDA + koffA;
    const __bf16* a1p = a0p + (size_t)16 * LDA;
    const __bf16* wp  = Wbase + (size_t)l15 * ldw + khB;
#pragma unroll
    for (int k0 = 0; k0 < K; k0 += 32) {
        bf16x8 a0lo = *(const bf16x8*)(a0p + k0);
        bf16x8 a0hi = *(const bf16x8*)(a0p + k0 + 8);
        bf16x8 a1lo = *(const bf16x8*)(a1p + k0);
        bf16x8 a1hi = *(const bf16x8*)(a1p + k0 + 8);
        v16bf a0 = __builtin_shufflevector(a0lo, a0hi, 0,1,2,3,4,5,6,7,8,9,10,11,12,13,14,15);
        v16bf a1 = __builtin_shufflevector(a1lo, a1hi, 0,1,2,3,4,5,6,7,8,9,10,11,12,13,14,15);
#pragma unroll
        for (int j = 0; j < 4; ++j) {
            v16bf b = *(const v16bf*)(wp + (size_t)(16 * j) * ldw + k0);
            acc[0][j] = __builtin_amdgcn_wmma_f32_16x16x32_bf16(
                false, a0, false, b, (short)0, acc[0][j], false, false);
            acc[1][j] = __builtin_amdgcn_wmma_f32_16x16x32_bf16(
                false, a1, false, b, (short)0, acc[1][j], false, false);
        }
    }
}

// ---------------------------------------------------------------------------
// Init: copy state into d_out regions, make bf16 copies of h and x.
// ---------------------------------------------------------------------------
__global__ __launch_bounds__(256) void init_state_kernel(
    const float* __restrict__ x_in, const float* __restrict__ h_in,
    const float* __restrict__ a_in, const float* __restrict__ b_in,
    float* __restrict__ h, float* __restrict__ alpha, float* __restrict__ beta,
    __bf16* __restrict__ hb, __bf16* __restrict__ xb)
{
    int idx = blockIdx.x * 256 + threadIdx.x;
    float hv = h_in[idx];
    h[idx]     = hv;
    hb[idx]    = (__bf16)hv;
    alpha[idx] = a_in[idx];
    beta[idx]  = b_in[idx];
    if (idx < BATCH * INPUT) xb[idx] = (__bf16)x_in[idx];
}

// ---------------------------------------------------------------------------
// Prep: weights -> bf16 (split Wf into Wx | Wh), rs = sigmoid(r), scalars.
// ---------------------------------------------------------------------------
__global__ __launch_bounds__(256) void prep_kernel(
    const float* __restrict__ Wf, const float* __restrict__ Wm,
    const float* __restrict__ rp, const float* __restrict__ tau,
    __bf16* __restrict__ Wxb, __bf16* __restrict__ Whb, __bf16* __restrict__ Wmb,
    float* __restrict__ rs, float* __restrict__ scal, float* __restrict__ sums,
    int* __restrict__ done, int* __restrict__ last)
{
    int idx = blockIdx.x * 256 + threadIdx.x;
    if (idx < HIDDEN * (INPUT + HIDDEN)) {
        int n = idx / (INPUT + HIDDEN);
        int k = idx % (INPUT + HIDDEN);
        __bf16 v = (__bf16)Wf[idx];
        if (k < INPUT) Wxb[n * INPUT + k] = v;
        else           Whb[n * HIDDEN + (k - INPUT)] = v;
    }
    if (idx < HIDDEN * NSYNCN) Wmb[idx] = (__bf16)Wm[idx];
    if (idx < NSYNCN) rs[idx] = 1.0f / (1.0f + __expf(-rp[idx]));
    if (idx == 0) {
        float t  = tau[0];
        float sp = (t > 20.0f) ? t : log1pf(__expf(t));   // softplus
        scal[0]  = 1.0f / sp;                             // inv_tau
        for (int i = 0; i < UNFOLDS; ++i) sums[i] = 0.0f;
        *done = 0;
        *last = 0;
    }
}

// ---------------------------------------------------------------------------
// Per-step synapse state: pair gather, alpha/beta EMA, sync -> bf16.
// ---------------------------------------------------------------------------
__global__ __launch_bounds__(256) void sync_kernel(
    const float* __restrict__ h, float* __restrict__ alpha, float* __restrict__ beta,
    const float* __restrict__ rs, const int* __restrict__ idxL, const int* __restrict__ idxR,
    __bf16* __restrict__ syncb, const int* __restrict__ done)
{
    if (*done) return;
    int idx = blockIdx.x * 256 + threadIdx.x;   // over B * NSYNC
    int b = idx >> 8;
    int j = idx & 255;
    float rsj = rs[j];
    size_t hrow = (size_t)b * HIDDEN;
    float pair = h[hrow + idxL[j]] * h[hrow + idxR[j]];
    float na = rsj * alpha[idx] + pair;
    float nb = rsj * beta[idx] + 1.0f;
    alpha[idx] = na;
    beta[idx]  = nb;
    syncb[idx] = (__bf16)(na / (sqrtf(nb) + 1e-6f));
}

// ---------------------------------------------------------------------------
// Fused GEMM + epilogue. Block = 8 waves = 64 rows x 256 cols.
// A tiles staged once per block into LDS by the Tensor Data Mover; all 4
// nwave-groups reuse them (4x fewer activation reads vs global-direct).
// ---------------------------------------------------------------------------
__global__ __launch_bounds__(256) void step_gemm_kernel(
    const __bf16* __restrict__ xb, const __bf16* __restrict__ hb,
    const __bf16* __restrict__ syncb,
    const __bf16* __restrict__ Wxb, const __bf16* __restrict__ Whb,
    const __bf16* __restrict__ Wmb,
    const float* __restrict__ bias, const float* __restrict__ Aparam,
    const float* __restrict__ h, const float* __restrict__ scal,
    float* __restrict__ hu, float* __restrict__ sum_slot,
    const int* __restrict__ done)
{
    if (*done) return;

    __shared__ __align__(16) __bf16 xs[64 * LDSX];   // 17 KB
    __shared__ __align__(16) __bf16 hs[64 * LDSH];   // 33 KB
    __shared__ __align__(16) __bf16 ss[64 * LDSH];   // 33 KB
    __shared__ float red[256];

    const int m0blk = blockIdx.x * 64;

    // Wave 0 issues the three TDM descriptors and waits on TENSORcnt.
    if (threadIdx.x < 32) {
        tdm_load_tile(xs, xb    + (size_t)m0blk * INPUT,  INPUT,  5u); // 64-DWORD rows
        tdm_load_tile(hs, hb    + (size_t)m0blk * HIDDEN, HIDDEN, 6u); // 128-DWORD rows
        tdm_load_tile(ss, syncb + (size_t)m0blk * HIDDEN, HIDDEN, 6u);
        __builtin_amdgcn_s_wait_tensorcnt(0);
    }
    __syncthreads();

    const int lane  = threadIdx.x & 31;
    const int wave  = threadIdx.x >> 5;
    const int mwave = wave & 1;
    const int nwave = wave >> 1;
    const int mloc  = mwave * 32;                  // row within the 64-row LDS tile
    const int n0    = nwave * 64;                  // 64 cols per wave

    v8f acc[2][4];
#pragma unroll
    for (int ms = 0; ms < 2; ++ms)
#pragma unroll
        for (int j = 0; j < 4; ++j) acc[ms][j] = {};

    gemm_seg<INPUT,  LDSX>(xs + (size_t)mloc * LDSX, Wxb + (size_t)n0 * INPUT,  INPUT,  acc, lane);
    gemm_seg<HIDDEN, LDSH>(hs + (size_t)mloc * LDSH, Whb + (size_t)n0 * HIDDEN, HIDDEN, acc, lane);
    gemm_seg<HIDDEN, LDSH>(ss + (size_t)mloc * LDSH, Wmb + (size_t)n0 * HIDDEN, HIDDEN, acc, lane);

    const float inv_tau = scal[0];
    const int   rofs = (lane < 16) ? 0 : 8;        // C/D layout: upper lanes hold M+8
    float lsum = 0.0f;
#pragma unroll
    for (int ms = 0; ms < 2; ++ms) {
#pragma unroll
        for (int j = 0; j < 4; ++j) {
            int n = n0 + 16 * j + (lane & 15);
            float bfn = bias[n];
            float An  = Aparam[n];
#pragma unroll
            for (int r = 0; r < 8; ++r) {
                int row = m0blk + mloc + 16 * ms + rofs + r;
                size_t o = (size_t)row * HIDDEN + n;
                float t = acc[ms][j][r] + bfn;
                float s = __sinf(t);
                float f = s * s;
                float hv = h[o];
                float huv = 0.1f * (-hv * (inv_tau + f) + An * f);
                hu[o] = huv;
                lsum += fabsf(huv);
            }
        }
    }
    // block reduction of |hu| partials, one float atomic per block
    red[threadIdx.x] = lsum;
    __syncthreads();
#pragma unroll
    for (int off = 128; off > 0; off >>= 1) {
        if (threadIdx.x < off) red[threadIdx.x] += red[threadIdx.x + off];
        __syncthreads();
    }
    if (threadIdx.x == 0) atomicAdd(sum_slot, red[0]);
}

// ---------------------------------------------------------------------------
// Finalize: h += (brk ? 1 : 2) * hu ; refresh bf16 copy of h.
// ---------------------------------------------------------------------------
__global__ __launch_bounds__(256) void finalize_kernel(
    float* __restrict__ h, __bf16* __restrict__ hb, const float* __restrict__ hu,
    const float* __restrict__ sum, const int* __restrict__ done, int s)
{
    if (*done) return;
    int idx = blockIdx.x * 256 + threadIdx.x;
    float norm = (*sum) * INV_BH;
    float w = ((s >= 3) && (norm < 0.01f)) ? 1.0f : 2.0f;
    float hn = h[idx] + w * hu[idx];
    h[idx]  = hn;
    hb[idx] = (__bf16)hn;
}

// ---------------------------------------------------------------------------
// Flag update (1 thread): done |= brk, last tracking, emit last as float.
// ---------------------------------------------------------------------------
__global__ void flags_kernel(const float* __restrict__ sum, int* __restrict__ done,
                             int* __restrict__ last, float* __restrict__ lout, int s)
{
    if (!(*done)) {
        *last = s;
        float norm = (*sum) * INV_BH;
        if ((s >= 3) && (norm < 0.01f)) *done = 1;
    }
    *lout = (float)(*last);
}

// ---------------------------------------------------------------------------
extern "C" void kernel_launch(void* const* d_in, const int* in_sizes, int n_in,
                              void* d_out, int out_size, void* d_ws, size_t ws_size,
                              hipStream_t stream)
{
    (void)in_sizes; (void)n_in; (void)out_size; (void)ws_size;

    const float* x_in  = (const float*)d_in[0];
    const float* h_in  = (const float*)d_in[1];
    const float* a_in  = (const float*)d_in[2];
    const float* b_in  = (const float*)d_in[3];
    const float* Wf    = (const float*)d_in[4];
    const float* bf    = (const float*)d_in[5];
    const float* tau   = (const float*)d_in[6];
    const float* rp    = (const float*)d_in[7];
    const float* Ap    = (const float*)d_in[8];
    const float* Wm    = (const float*)d_in[9];
    const int*   idxL  = (const int*)d_in[10];
    const int*   idxR  = (const int*)d_in[11];

    // Outputs live in d_out and are updated in place: [h | alpha | beta | last]
    float* hout = (float*)d_out;
    float* aout = hout + BH;
    float* bout = aout + BH;
    float* lout = bout + BH;

    // Workspace layout (all offsets 32B-aligned)
    char* ws = (char*)d_ws;
    float*  hu    = (float*)ws;                               // 32 MB
    __bf16* xb    = (__bf16*)(ws + ((size_t)32 << 20));       //  8 MB
    __bf16* hb    = (__bf16*)(ws + ((size_t)40 << 20));       // 16 MB
    __bf16* syncb = (__bf16*)(ws + ((size_t)56 << 20));       // 16 MB
    char*   wsw   = ws + ((size_t)72 << 20);
    __bf16* Wxb   = (__bf16*)(wsw);                           // 64 KB
    __bf16* Whb   = (__bf16*)(wsw + 65536);                   // 128 KB
    __bf16* Wmb   = (__bf16*)(wsw + 65536 + 131072);          // 128 KB
    float*  rs    = (float*)(wsw + 65536 + 131072 * 2);       // 1 KB
    float*  scal  = rs + NSYNCN;                              // [0] = inv_tau
    float*  sums  = scal + 8;                                 // 6 step sums
    int*    done  = (int*)(sums + 8);
    int*    last  = done + 1;

    init_state_kernel<<<BH / 256, 256, 0, stream>>>(
        x_in, h_in, a_in, b_in, hout, aout, bout, hb, xb);
    prep_kernel<<<(HIDDEN * (INPUT + HIDDEN)) / 256, 256, 0, stream>>>(
        Wf, Wm, rp, tau, Wxb, Whb, Wmb, rs, scal, sums, done, last);

    for (int s = 0; s < UNFOLDS; ++s) {
        sync_kernel<<<BH / 256, 256, 0, stream>>>(
            hout, aout, bout, rs, idxL, idxR, syncb, done);
        step_gemm_kernel<<<BATCH / 64, 256, 0, stream>>>(
            xb, hb, syncb, Wxb, Whb, Wmb, bf, Ap, hout, scal, hu, &sums[s], done);
        finalize_kernel<<<BH / 256, 256, 0, stream>>>(
            hout, hb, hu, &sums[s], done, s);
        flags_kernel<<<1, 1, 0, stream>>>(&sums[s], done, last, lout, s);
    }
}
